// EmbedderGNNv1_85555748536460
// MI455X (gfx1250) — matrix operand order repaired and checked by reading, not compiled, not run on hardware
//
#include <hip/hip_runtime.h>
#include <hip/hip_bf16.h>
#include <stdint.h>

// ---------------------------------------------------------------------------
// GraphSAGE 2-layer forward for MI455X (gfx1250, wave32, WMMA).
// GEMMs use v_wmma_f32_16x16x32_bf16 with bf16 hi/lo split precision
// (3 WMMAs per tile) to approximate fp32 accuracy at WMMA throughput.
// B (weight) tiles staged via async global->LDS (ASYNCcnt path).
// ---------------------------------------------------------------------------

typedef __attribute__((ext_vector_type(16))) __bf16 v16bf;
typedef __attribute__((ext_vector_type(8)))  float  v8f;

#define DFEAT 512
#define KTOT  1024
#define BM    128
#define BN    128
#define KC    32

#if defined(__has_builtin)
#if __has_builtin(__builtin_amdgcn_global_load_async_to_lds_b128) && \
    __has_builtin(__builtin_amdgcn_s_wait_asynccnt)
#define USE_ASYNC_LDS 1
#endif
#endif

#ifdef USE_ASYNC_LDS
typedef int v4i_async __attribute__((vector_size(16)));
typedef __attribute__((address_space(1))) v4i_async* async_gptr;
typedef __attribute__((address_space(3))) v4i_async* async_lptr;
#endif

__device__ __forceinline__ unsigned short f2bf(float f) {
  union { float f; unsigned u; } v; v.f = f;
  unsigned r = v.u + 0x7FFFu + ((v.u >> 16) & 1u);   // round to nearest even
  return (unsigned short)(r >> 16);
}
__device__ __forceinline__ float bf2f(unsigned short h) {
  union { unsigned u; float f; } v; v.u = ((unsigned)h) << 16;
  return v.f;
}

// ---------------------------------------------------------------------------
// Edge scatter: agg[dst] += x[src], deg[dst] += 1   (one block per edge)
// ---------------------------------------------------------------------------
__global__ __launch_bounds__(256)
void scatter_accum(const float* __restrict__ x,
                   const int* __restrict__ src,
                   const int* __restrict__ dst,
                   float* __restrict__ agg,
                   float* __restrict__ deg, int nedges) {
  int e = blockIdx.x;
  if (e >= nedges) return;
  int s = src[e], d = dst[e];
  const float* xs = x + (size_t)s * DFEAT;
  float*       ad = agg + (size_t)d * DFEAT;
  int f = threadIdx.x * 2;
  float2 v = *(const float2*)(xs + f);
  __hip_atomic_fetch_add(ad + f,     v.x, __ATOMIC_RELAXED, __HIP_MEMORY_SCOPE_AGENT);
  __hip_atomic_fetch_add(ad + f + 1, v.y, __ATOMIC_RELAXED, __HIP_MEMORY_SCOPE_AGENT);
  if (threadIdx.x == 0)
    __hip_atomic_fetch_add(deg + d, 1.0f, __ATOMIC_RELAXED, __HIP_MEMORY_SCOPE_AGENT);
}

// agg[n][:] /= max(deg[n],1)
__global__ __launch_bounds__(256)
void normalize_rows(float* __restrict__ agg, const float* __restrict__ deg, int total) {
  int i = blockIdx.x * blockDim.x + threadIdx.x;
  if (i >= total) return;
  int row = i >> 9;                       // /512
  float d = fmaxf(deg[row], 1.0f);
  agg[i] = agg[i] / d;
}

// ---------------------------------------------------------------------------
// Pre-transpose weights into B layout: B[k][o] = (k<512 ? Wl[o][k] : Wr[o][k-512])
// stored as bf16 hi + lo planes ([1024][512]).
// ---------------------------------------------------------------------------
__global__ __launch_bounds__(256)
void build_b(const float* __restrict__ Wl, const float* __restrict__ Wr,
             unsigned short* __restrict__ Bh, unsigned short* __restrict__ Bl) {
  int idx = blockIdx.x * blockDim.x + threadIdx.x;
  if (idx >= KTOT * DFEAT) return;
  int k = idx >> 9;
  int o = idx & 511;
  float w = (k < DFEAT) ? Wl[(size_t)o * DFEAT + k]
                        : Wr[(size_t)o * DFEAT + (k - DFEAT)];
  unsigned short hi = f2bf(w);
  unsigned short lo = f2bf(w - bf2f(hi));
  Bh[idx] = hi;
  Bl[idx] = lo;
}

// ---------------------------------------------------------------------------
// Fused SAGE GEMM: C = [A1 | A2] (N x 1024) @ B (1024 x 512) + bias
//   A1 = normalized aggregate, A2 = node features; B pre-converted bf16 hi/lo.
// Block: 256 threads (8 wave32), 128x128 output tile; K staged in LDS by 32.
// Split precision: acc += Ahi*Bhi + Alo*Bhi + Ahi*Blo.
// ---------------------------------------------------------------------------
__global__ __launch_bounds__(256)
void sage_gemm(const float* __restrict__ A1,
               const float* __restrict__ A2,
               const unsigned short* __restrict__ Bhi,
               const unsigned short* __restrict__ Blo,
               const float* __restrict__ bias,
               float* __restrict__ C, int nrows) {
  __shared__ unsigned short As_hi[BM][KC];
  __shared__ unsigned short As_lo[BM][KC];
  __shared__ unsigned short Bs_hi[KC][BN];
  __shared__ unsigned short Bs_lo[KC][BN];

  const int tid  = threadIdx.x;
  const int lane = tid & 31;
  const int wave = tid >> 5;
  const int wm   = wave >> 2;      // 0..1  -> 64 rows each
  const int wn   = wave & 3;       // 0..3  -> 32 cols each
  const int brow = blockIdx.x * BM;
  const int bcol = blockIdx.y * BN;

  v8f acc[4][2];
#pragma unroll
  for (int i = 0; i < 4; ++i)
#pragma unroll
    for (int j = 0; j < 2; ++j) { v8f z = {}; acc[i][j] = z; }

  union Frag { unsigned u[8]; v16bf v; };

  for (int k0 = 0; k0 < KTOT; k0 += KC) {
    const float* Asrc = (k0 < DFEAT) ? A1 : A2;
    const int kc = (k0 < DFEAT) ? k0 : (k0 - DFEAT);

    __syncthreads();
    // ---- stage A chunk (128x32 fp32 -> bf16 hi/lo) : 1024 float4, 4/thread
#pragma unroll
    for (int i = 0; i < 4; ++i) {
      int idx = tid + 256 * i;
      int r = idx >> 3;            // row in tile
      int q = idx & 7;             // float4 within the 32-wide chunk
      int grow = brow + r;
      float4 v = make_float4(0.f, 0.f, 0.f, 0.f);
      if (grow < nrows)
        v = *(const float4*)(Asrc + (size_t)grow * DFEAT + kc + q * 4);
      float vv[4] = {v.x, v.y, v.z, v.w};
#pragma unroll
      for (int j = 0; j < 4; ++j) {
        unsigned short hi = f2bf(vv[j]);
        unsigned short lo = f2bf(vv[j] - bf2f(hi));
        As_hi[r][q * 4 + j] = hi;
        As_lo[r][q * 4 + j] = lo;
      }
    }
    // ---- stage B chunk (32x128 bf16 hi/lo) : 512 x b128 per plane, 2/thread
#pragma unroll
    for (int i = 0; i < 2; ++i) {
      int idx = tid + 256 * i;     // 0..511
      int kr  = idx >> 4;          // 0..31
      int cq  = idx & 15;          // b128 (8 ushorts) within 128-wide row
      size_t gb = (size_t)(k0 + kr) * DFEAT + bcol + cq * 8;
#ifdef USE_ASYNC_LDS
      // CDNA5 async copy: bypasses VGPRs, tracked by ASYNCcnt
      __builtin_amdgcn_global_load_async_to_lds_b128(
          (async_gptr)(uintptr_t)(Bhi + gb),
          (async_lptr)&Bs_hi[kr][cq * 8], 0, 0);
      __builtin_amdgcn_global_load_async_to_lds_b128(
          (async_gptr)(uintptr_t)(Blo + gb),
          (async_lptr)&Bs_lo[kr][cq * 8], 0, 0);
#else
      *(uint4*)&Bs_hi[kr][cq * 8] = *(const uint4*)(Bhi + gb);
      *(uint4*)&Bs_lo[kr][cq * 8] = *(const uint4*)(Blo + gb);
#endif
    }
#ifdef USE_ASYNC_LDS
    __builtin_amdgcn_s_wait_asynccnt(0);
#endif
    __syncthreads();

    // ---- prefetch next chunk's A rows into cache while we compute
    if (k0 + KC < KTOT) {
      const float* An = ((k0 + KC) < DFEAT) ? A1 : A2;
      const int kn = ((k0 + KC) < DFEAT) ? (k0 + KC) : (k0 + KC - DFEAT);
      int r = tid >> 3;
      int q = tid & 7;
      int grow = brow + r;
      if (grow < nrows)
        __builtin_prefetch(An + (size_t)grow * DFEAT + kn + q * 4, 0, 1);
    }

    // ---- B fragments (col n = lane%16, k pairs in table order per VGPR, matching A)
    Frag bh[2], bl[2];
#pragma unroll
    for (int sn = 0; sn < 2; ++sn) {
      int col = wn * 32 + sn * 16;
#pragma unroll
      for (int v = 0; v < 8; ++v) {
        int kk = (v >> 2) * 16 + ((lane >> 4) << 3) + (v & 3) * 2;
        int c = col + (lane & 15);
        bh[sn].u[v] = (unsigned)Bs_hi[kk][c] | ((unsigned)Bs_hi[kk + 1][c] << 16);
        bl[sn].u[v] = (unsigned)Bs_lo[kk][c] | ((unsigned)Bs_lo[kk + 1][c] << 16);
      }
    }
    // ---- A fragments + WMMA
#pragma unroll
    for (int sm = 0; sm < 4; ++sm) {
      int row = wm * 64 + sm * 16 + (lane & 15);
      Frag ah, al;
#pragma unroll
      for (int v = 0; v < 8; ++v) {
        int kk = (v >> 2) * 16 + ((lane >> 4) << 3) + (v & 3) * 2;
        ah.u[v] = *(const unsigned*)&As_hi[row][kk];
        al.u[v] = *(const unsigned*)&As_lo[row][kk];
      }
#pragma unroll
      for (int sn = 0; sn < 2; ++sn) {
        acc[sm][sn] = __builtin_amdgcn_wmma_f32_16x16x32_bf16(
            false, ah.v, false, bh[sn].v, (short)0, acc[sm][sn], false, false);
        acc[sm][sn] = __builtin_amdgcn_wmma_f32_16x16x32_bf16(
            false, al.v, false, bh[sn].v, (short)0, acc[sm][sn], false, false);
        acc[sm][sn] = __builtin_amdgcn_wmma_f32_16x16x32_bf16(
            false, ah.v, false, bl[sn].v, (short)0, acc[sm][sn], false, false);
      }
    }
  }

  // ---- epilogue: C-layout m = j + 8*(lane/16), n = lane%16; add bias
  int mofs = (lane >> 4) << 3;
  int nloc = lane & 15;
#pragma unroll
  for (int sm = 0; sm < 4; ++sm)
#pragma unroll
    for (int sn = 0; sn < 2; ++sn) {
      int gcol = bcol + wn * 32 + sn * 16 + nloc;
      float bv = bias[gcol];
#pragma unroll
      for (int j = 0; j < 8; ++j) {
        int grow = brow + wm * 64 + sm * 16 + mofs + j;
        if (grow < nrows)
          C[(size_t)grow * DFEAT + gcol] = acc[sm][sn][j] + bv;
      }
    }
}

// ---------------------------------------------------------------------------
// Fused LayerNorm (+optional ReLU) + residual: one block per row.
// ---------------------------------------------------------------------------
__global__ __launch_bounds__(256)
void layernorm_fuse(const float* __restrict__ in, const float* __restrict__ res,
                    const float* __restrict__ w, const float* __restrict__ b,
                    float* __restrict__ out, int do_relu) {
  __shared__ float red[256];
  int row = blockIdx.x;
  int t = threadIdx.x;
  const float* pin  = in  + (size_t)row * DFEAT;
  const float* pres = res + (size_t)row * DFEAT;
  float*       pout = out + (size_t)row * DFEAT;

  float2 v = *(const float2*)(pin + t * 2);
  red[t] = v.x + v.y;
  __syncthreads();
#pragma unroll
  for (int off = 128; off > 0; off >>= 1) {
    if (t < off) red[t] += red[t + off];
    __syncthreads();
  }
  float mu = red[0] * (1.0f / DFEAT);
  __syncthreads();

  float dx = v.x - mu, dy = v.y - mu;
  red[t] = dx * dx + dy * dy;
  __syncthreads();
#pragma unroll
  for (int off = 128; off > 0; off >>= 1) {
    if (t < off) red[t] += red[t + off];
    __syncthreads();
  }
  float var = red[0] * (1.0f / DFEAT);
  float rstd = rsqrtf(var + 1e-5f);

  float2 wv = *(const float2*)(w + t * 2);
  float2 bv = *(const float2*)(b + t * 2);
  float2 rv = *(const float2*)(pres + t * 2);
  float o0 = dx * rstd * wv.x + bv.x;
  float o1 = dy * rstd * wv.y + bv.y;
  if (do_relu) { o0 = fmaxf(o0, 0.f); o1 = fmaxf(o1, 0.f); }
  pout[t * 2]     = o0 + rv.x;
  pout[t * 2 + 1] = o1 + rv.y;
}

// ---------------------------------------------------------------------------
extern "C" void kernel_launch(void* const* d_in, const int* in_sizes, int n_in,
                              void* d_out, int out_size, void* d_ws, size_t ws_size,
                              hipStream_t stream) {
  const float* x    = (const float*)d_in[0];
  const int*   ei   = (const int*)d_in[1];
  const float* W1l  = (const float*)d_in[2];
  const float* b1l  = (const float*)d_in[3];
  const float* W1r  = (const float*)d_in[4];
  const float* W2l  = (const float*)d_in[5];
  const float* b2l  = (const float*)d_in[6];
  const float* W2r  = (const float*)d_in[7];
  const float* ln1w = (const float*)d_in[8];
  const float* ln1b = (const float*)d_in[9];
  const float* ln2w = (const float*)d_in[10];
  const float* ln2b = (const float*)d_in[11];
  float* out = (float*)d_out;

  const int N = in_sizes[0] / DFEAT;   // 50000
  const int E = in_sizes[1] / 2;       // 400000
  const int* src = ei;
  const int* dst = ei + E;

  // workspace carve-out
  char* ws = (char*)d_ws;
  size_t off = 0;
  auto carve = [&](size_t bytes) -> void* {
    void* p = ws + off;
    off = (off + bytes + 255) & ~(size_t)255;
    return p;
  };
  const size_t fbytes = (size_t)N * DFEAT * sizeof(float);
  float* agg = (float*)carve(fbytes);
  float* tmp = (float*)carve(fbytes);
  float* h   = (float*)carve(fbytes);
  float* deg = (float*)carve((size_t)N * sizeof(float));
  unsigned short* B1h = (unsigned short*)carve((size_t)KTOT * DFEAT * 2);
  unsigned short* B1l = (unsigned short*)carve((size_t)KTOT * DFEAT * 2);
  unsigned short* B2h = (unsigned short*)carve((size_t)KTOT * DFEAT * 2);
  unsigned short* B2l = (unsigned short*)carve((size_t)KTOT * DFEAT * 2);

  const int total = N * DFEAT;
  dim3 gemmGrid((N + BM - 1) / BM, DFEAT / BN);

  // weight pre-transpose + bf16 split (both layers)
  build_b<<<(KTOT * DFEAT + 255) / 256, 256, 0, stream>>>(W1l, W1r, B1h, B1l);
  build_b<<<(KTOT * DFEAT + 255) / 256, 256, 0, stream>>>(W2l, W2r, B2h, B2l);

  // ---- layer 1 ----
  (void)hipMemsetAsync(agg, 0, fbytes, stream);
  (void)hipMemsetAsync(deg, 0, (size_t)N * sizeof(float), stream);
  scatter_accum<<<E, 256, 0, stream>>>(x, src, dst, agg, deg, E);
  normalize_rows<<<(total + 255) / 256, 256, 0, stream>>>(agg, deg, total);
  sage_gemm<<<gemmGrid, 256, 0, stream>>>(agg, x, B1h, B1l, b1l, tmp, N);
  layernorm_fuse<<<N, 256, 0, stream>>>(tmp, x, ln1w, ln1b, h, /*relu=*/1);

  // ---- layer 2 ----
  (void)hipMemsetAsync(agg, 0, fbytes, stream);
  (void)hipMemsetAsync(deg, 0, (size_t)N * sizeof(float), stream);
  scatter_accum<<<E, 256, 0, stream>>>(h, src, dst, agg, deg, E);
  normalize_rows<<<(total + 255) / 256, 256, 0, stream>>>(agg, deg, total);
  sage_gemm<<<gemmGrid, 256, 0, stream>>>(agg, h, B2h, B2l, b2l, tmp, N);
  layernorm_fuse<<<N, 256, 0, stream>>>(tmp, h, ln2w, ln2b, out, /*relu=*/0);

  (void)n_in; (void)out_size; (void)ws_size;
}
